// WhileOpGridLSTMNet_47399259079398
// MI455X (gfx1250) — matrix operand order repaired and checked by reading, not compiled, not run on hardware
//
#include <hip/hip_runtime.h>
#include <hip/hip_bf16.h>

// Grid-RNN (WhileOpGridLSTMNet): DEPTH=2, SRC=32, TRG=32, B=32, H=256
// Wavefront over anti-diagonals; one workgroup (256 thr = 8 waves) per cell.
// Per cell: [32 x 1024] (f16, LDS) x [1024-ish x 256] (f16, L2) via
// v_wmma_f32_16x16x32_f16, f32 accumulate, tanh epilogue, f32 store.

#define DEPTHD 2
#define SRCL   32
#define TRGL   32
#define BATCH  32
#define HID    256
#define KA     1024   // LDS A columns: sx(256)|prev_hx(256)|x_in(256)|y_in(256)
#define KW     768    // fused weight rows: Ux(256)|Uy(256)|W(256)

typedef __attribute__((ext_vector_type(16))) _Float16 v16h;
typedef __attribute__((ext_vector_type(8)))  _Float16 v8h;
typedef __attribute__((ext_vector_type(8)))  float    v8f;

union AFrag { v16h v; v8h h[2]; };

// start of the [B,H] block for output (d,i,j,c) in [DEPTH,SRC,TRG,2,B,H] f32
__device__ __forceinline__ size_t ooff(int d, int i, int j, int c) {
    return ((size_t)(((d * SRCL + i) * TRGL + j) * 2 + c)) * (BATCH * HID);
}

// Build fused, transposed f16 weights: Wt[d][n][k], k<512 -> U[d][k][n], else W[d][k-512][n]
__global__ __launch_bounds__(256) void prep_weights_kernel(
    const float* __restrict__ W,   // [DEPTH, 256, 256]
    const float* __restrict__ U,   // [DEPTH, 512, 256]
    _Float16* __restrict__ Wt)     // [DEPTH, 256, 768]
{
    int idx = blockIdx.x * blockDim.x + threadIdx.x;
    if (idx >= DEPTHD * HID * KW) return;
    int k = idx % KW;
    int n = (idx / KW) % HID;
    int d = idx / (KW * HID);
    float v = (k < 512) ? U[((size_t)d * 512 + k) * HID + n]
                        : W[((size_t)d * 256 + (k - 512)) * HID + n];
    Wt[idx] = (_Float16)v;
}

__device__ __forceinline__ void fill_block(_Float16* smA, const float* __restrict__ ptr,
                                           int q, int row, int c0, bool valid)
{
    _Float16* dst = smA + row * KA + q * 256 + c0;
    if (valid) {
        const float4* p = (const float4*)(ptr + row * HID + c0);
#pragma unroll
        for (int t = 0; t < 8; ++t) {
            float4 v = p[t];
            dst[4 * t + 0] = (_Float16)v.x;
            dst[4 * t + 1] = (_Float16)v.y;
            dst[4 * t + 2] = (_Float16)v.z;
            dst[4 * t + 3] = (_Float16)v.w;
        }
    } else {
#pragma unroll
        for (int t = 0; t < 32; ++t) dst[t] = (_Float16)0.0f;
    }
}

__global__ __launch_bounds__(256) void grid_cell_kernel(
    const float* __restrict__ src,   // [SRC, B, H]
    const float* __restrict__ trg,   // [TRG, B, H]
    const float* __restrict__ bias,  // [DEPTH, 1, H]
    const _Float16* __restrict__ Wt, // [DEPTH, 256(N), 768(K)]
    float* __restrict__ out,         // [DEPTH, SRC, TRG, 2, B, H]
    int d, int diag)
{
    __shared__ _Float16 smA[BATCH * KA];   // 64 KB: states + inputs, f16

    const int i0 = (diag > TRGL - 1) ? diag - (TRGL - 1) : 0;
    const int i  = i0 + blockIdx.x;
    const int j  = diag - i;
    const int tid = threadIdx.x;

    // ---- stage A panel into LDS (f32 -> f16) -------------------------------
    const int row = tid >> 3;          // 0..31 (batch row)
    const int c0  = (tid & 7) * 32;    // 0..224 (col chunk of 32)

    const float* p_sx = out + ooff(d, (i > 0) ? i - 1 : 0, j, 0);
    const float* p_ph = out + ooff(d, i, (j > 0) ? j - 1 : 0, 0);
    const float* p_x  = (d == 0) ? (src + (size_t)i * BATCH * HID) : (out + ooff(d - 1, i, j, 0));
    const float* p_y  = (d == 0) ? (trg + (size_t)j * BATCH * HID) : (out + ooff(d - 1, i, j, 1));

    fill_block(smA, p_sx, 0, row, c0, i > 0);
    fill_block(smA, p_ph, 1, row, c0, j > 0);
    fill_block(smA, p_x,  2, row, c0, true);
    fill_block(smA, p_y,  3, row, c0, true);
    __syncthreads();

    // ---- WMMA compute: each wave owns M=32 x N=32 output tile --------------
    const int wave = tid >> 5;
    const int lane = tid & 31;
    const int lh   = lane & 15;
    const int akb  = (lane < 16) ? 0 : 16;   // A-frag K sub-offset (contiguous K, matches B)
    const int bkb  = (lane < 16) ? 0 : 16;   // B-frag K sub-offset (contiguous K, matches A)
    const int n0   = wave * 32 + lh;         // first N column (nt=0); nt=1 -> +16

    const _Float16* wb0 = Wt + ((size_t)d * HID + n0) * KW;
    const _Float16* wb1 = wb0 + (size_t)16 * KW;

    v8f acc_t[2][2] = {};

    // shared recurrent part: K = 0..511  (sx@Ux + prev_hx@Uy)
    for (int kc = 0; kc < 16; ++kc) {
        const int ka = kc * 32;
        v16h bf0 = *(const v16h*)(wb0 + ka + bkb);
        v16h bf1 = *(const v16h*)(wb1 + ka + bkb);
        AFrag af[2];
#pragma unroll
        for (int mt = 0; mt < 2; ++mt) {
            const _Float16* ap = smA + (size_t)(lh + 16 * mt) * KA + ka + akb;
            af[mt].h[0] = *(const v8h*)(ap);
            af[mt].h[1] = *(const v8h*)(ap + 8);
        }
#pragma unroll
        for (int mt = 0; mt < 2; ++mt) {
            acc_t[mt][0] = __builtin_amdgcn_wmma_f32_16x16x32_f16(
                false, af[mt].v, false, bf0, (short)0, acc_t[mt][0], false, false);
            acc_t[mt][1] = __builtin_amdgcn_wmma_f32_16x16x32_f16(
                false, af[mt].v, false, bf1, (short)0, acc_t[mt][1], false, false);
        }
    }

    v8f acc_x[2][2], acc_y[2][2];
#pragma unroll
    for (int mt = 0; mt < 2; ++mt)
#pragma unroll
        for (int nt = 0; nt < 2; ++nt) { acc_x[mt][nt] = acc_t[mt][nt]; acc_y[mt][nt] = acc_t[mt][nt]; }

    // input projections: weight rows 512..767 (W), A regions x_in / y_in
    for (int kc = 0; kc < 8; ++kc) {
        const int kw = 512 + kc * 32;
        v16h bf0 = *(const v16h*)(wb0 + kw + bkb);
        v16h bf1 = *(const v16h*)(wb1 + kw + bkb);
        AFrag ax[2], ay[2];
#pragma unroll
        for (int mt = 0; mt < 2; ++mt) {
            const _Float16* apx = smA + (size_t)(lh + 16 * mt) * KA + (512 + kc * 32) + akb;
            const _Float16* apy = smA + (size_t)(lh + 16 * mt) * KA + (768 + kc * 32) + akb;
            ax[mt].h[0] = *(const v8h*)(apx);  ax[mt].h[1] = *(const v8h*)(apx + 8);
            ay[mt].h[0] = *(const v8h*)(apy);  ay[mt].h[1] = *(const v8h*)(apy + 8);
        }
#pragma unroll
        for (int mt = 0; mt < 2; ++mt) {
            acc_x[mt][0] = __builtin_amdgcn_wmma_f32_16x16x32_f16(
                false, ax[mt].v, false, bf0, (short)0, acc_x[mt][0], false, false);
            acc_x[mt][1] = __builtin_amdgcn_wmma_f32_16x16x32_f16(
                false, ax[mt].v, false, bf1, (short)0, acc_x[mt][1], false, false);
            acc_y[mt][0] = __builtin_amdgcn_wmma_f32_16x16x32_f16(
                false, ay[mt].v, false, bf0, (short)0, acc_y[mt][0], false, false);
            acc_y[mt][1] = __builtin_amdgcn_wmma_f32_16x16x32_f16(
                false, ay[mt].v, false, bf1, (short)0, acc_y[mt][1], false, false);
        }
    }

    // ---- epilogue: bias + tanh, f32 stores ---------------------------------
    const float bz0 = bias[(size_t)d * HID + n0];
    const float bz1 = bias[(size_t)d * HID + n0 + 16];
    const size_t bx = ooff(d, i, j, 0);
    const size_t by = ooff(d, i, j, 1);
    const int mrb = 8 * (lane >> 4);   // C/D layout: lanes 16-31 hold M+8 rows

#pragma unroll
    for (int mt = 0; mt < 2; ++mt) {
#pragma unroll
        for (int nt = 0; nt < 2; ++nt) {
            const int n = n0 + 16 * nt;
            const float bv = (nt == 0) ? bz0 : bz1;
#pragma unroll
            for (int r = 0; r < 8; ++r) {
                const int m = 16 * mt + mrb + r;
                out[bx + (size_t)m * HID + n] = tanhf(acc_x[mt][nt][r] + bv);
                out[by + (size_t)m * HID + n] = tanhf(acc_y[mt][nt][r] + bv);
            }
        }
    }
}

extern "C" void kernel_launch(void* const* d_in, const int* in_sizes, int n_in,
                              void* d_out, int out_size, void* d_ws, size_t ws_size,
                              hipStream_t stream) {
    const float* src = (const float*)d_in[0];   // [32, 32, 256]
    const float* trg = (const float*)d_in[1];   // [32, 32, 256]
    const float* W   = (const float*)d_in[2];   // [2, 256, 256]
    const float* U   = (const float*)d_in[3];   // [2, 512, 256]
    const float* b   = (const float*)d_in[4];   // [2, 1, 256]
    float* out = (float*)d_out;                 // [2, 32, 32, 2, 32, 256]
    _Float16* Wt = (_Float16*)d_ws;             // [2, 256, 768] f16 (786 KB)

    {
        const int total = DEPTHD * HID * KW;
        prep_weights_kernel<<<(total + 255) / 256, 256, 0, stream>>>(W, U, Wt);
    }

    for (int d = 0; d < DEPTHD; ++d) {
        for (int s = 0; s <= SRCL + TRGL - 2; ++s) {
            int cnt = SRCL - ((s > SRCL - 1) ? s - (SRCL - 1) : (SRCL - 1) - s);
            grid_cell_kernel<<<cnt, 256, 0, stream>>>(src, trg, b, Wt, out, d, s);
        }
    }
}
